// Shakespeare_GPT_83227876262562
// MI455X (gfx1250) — matrix operand order repaired and checked by reading, not compiled, not run on hardware
//
#include <hip/hip_runtime.h>

#define BB 2
#define TT 1024
#define CC 1024
#define HH 16
#define LL 6
#define VV 32000
#define DDIM 64
#define MM (BB*TT)          // 2048 rows

typedef __attribute__((ext_vector_type(16))) __bf16 v16bf;
typedef __attribute__((ext_vector_type(8)))  float  v8f;

#if defined(__has_builtin)
#if __has_builtin(__builtin_amdgcn_global_load_async_to_lds_b128)
#define HAVE_ASYNC 1
#endif
#endif
#ifndef HAVE_ASYNC
#define HAVE_ASYNC 0
#endif

__device__ __forceinline__ unsigned short f2bf(float f) {
    unsigned int u = __float_as_uint(f);
    u += 0x7FFFu + ((u >> 16) & 1u);          // round-to-nearest-even
    return (unsigned short)(u >> 16);
}

// 16-byte global -> LDS copy (async on CDNA5, tracked by ASYNCcnt)
typedef int v4i_vs __attribute__((vector_size(16)));
typedef __attribute__((address_space(1))) v4i_vs* gvec128_t;
typedef __attribute__((address_space(3))) v4i_vs* lvec128_t;

__device__ __forceinline__ void copy16_g2l(const unsigned short* g, unsigned short* l) {
#if HAVE_ASYNC
    __builtin_amdgcn_global_load_async_to_lds_b128((gvec128_t)g, (lvec128_t)l, 0, 0);
#else
    *(uint4*)l = *(const uint4*)g;
#endif
}

__device__ __forceinline__ void wait_g2l() {
#if HAVE_ASYNC
#if __has_builtin(__builtin_amdgcn_s_wait_asynccnt)
    __builtin_amdgcn_s_wait_asynccnt(0);
#else
    asm volatile("s_wait_asynccnt 0" ::: "memory");
#endif
#endif
}

// ---------------- weight convert + transpose: fp32 [K,N] -> bf16 [N,K] -------
__global__ void k_convT(const float* __restrict__ in,
                        unsigned short* __restrict__ out,
                        int K, int N) {
    long long i = (long long)blockIdx.x * blockDim.x + threadIdx.x;
    long long total = (long long)K * N;
    if (i >= total) return;
    int k = (int)(i / N);
    int n = (int)(i % N);
    out[(long long)n * K + k] = f2bf(in[i]);
}

// ---------------- embedding: x = tok_emb[idx] + pos_emb ----------------------
__global__ void k_embed(const int* __restrict__ idx,
                        const float* __restrict__ tok,
                        const float* __restrict__ pos,
                        float* __restrict__ x) {
    int i = blockIdx.x * blockDim.x + threadIdx.x;   // < MM*CC
    int m = i / CC, c = i % CC;
    int t = m % TT;
    x[i] = tok[(long long)idx[m] * CC + c] + pos[t * CC + c];
}

// ---------------- layernorm (row per block), emits bf16 ----------------------
__global__ void __launch_bounds__(256)
k_ln(const float* __restrict__ x, const float* __restrict__ g,
     const float* __restrict__ b, unsigned short* __restrict__ out) {
    int row = blockIdx.x;
    const float* xr = x + (long long)row * CC;
    __shared__ float red[256];
    float v[4];
    float s = 0.f;
    #pragma unroll
    for (int i = 0; i < 4; i++) { v[i] = xr[threadIdx.x + i * 256]; s += v[i]; }
    red[threadIdx.x] = s; __syncthreads();
    for (int off = 128; off > 0; off >>= 1) {
        if (threadIdx.x < off) red[threadIdx.x] += red[threadIdx.x + off];
        __syncthreads();
    }
    float mu = red[0] * (1.f / CC);
    __syncthreads();
    float s2 = 0.f;
    #pragma unroll
    for (int i = 0; i < 4; i++) { float d = v[i] - mu; s2 += d * d; }
    red[threadIdx.x] = s2; __syncthreads();
    for (int off = 128; off > 0; off >>= 1) {
        if (threadIdx.x < off) red[threadIdx.x] += red[threadIdx.x + off];
        __syncthreads();
    }
    float rstd = rsqrtf(red[0] * (1.f / CC) + 1e-5f);
    #pragma unroll
    for (int i = 0; i < 4; i++) {
        int c = threadIdx.x + i * 256;
        out[(long long)row * CC + c] = f2bf((v[i] - mu) * rstd * g[c] + b[c]);
    }
}

// ---------------- bf16 WMMA GEMM, async-LDS tiling, double-buffered ----------
#define LDSTR 72

__global__ void __launch_bounds__(256)
k_gemm(const unsigned short* __restrict__ A,
       const unsigned short* __restrict__ Bt,
       const float* __restrict__ bias,
       const float* __restrict__ residual,
       float* __restrict__ outF,
       unsigned short* __restrict__ outB,
       int M, int N, int K, int relu) {
    __shared__ unsigned short Alds[2][64 * LDSTR];
    __shared__ unsigned short Blds[2][128 * LDSTR];

    int tid  = threadIdx.x;
    int lane = tid & 31;
    int wave = tid >> 5;
    int tileM = blockIdx.y * 64;
    int tileN = blockIdx.x * 128;
    int waveM = (wave & 3) * 16;
    int waveN = (wave >> 2) * 64;
    int l15   = lane & 15;
    int half  = (lane >> 4) & 1;

    int arow0 = tid >> 3,           acol = (tid & 7) * 8;
    int arow1 = (tid + 256) >> 3;

    v8f acc[4];
    v8f z = {0.f,0.f,0.f,0.f,0.f,0.f,0.f,0.f};
    #pragma unroll
    for (int i = 0; i < 4; i++) acc[i] = z;

    int nk = K >> 6;   // K / 64

    // prologue: stage tile 0 into buffer 0
    {
        copy16_g2l(A + (long long)(tileM + arow0) * K + acol, &Alds[0][arow0 * LDSTR + acol]);
        copy16_g2l(A + (long long)(tileM + arow1) * K + acol, &Alds[0][arow1 * LDSTR + acol]);
        #pragma unroll
        for (int c = 0; c < 4; c++) {
            int chunk = tid + c * 256;
            int row = chunk >> 3, col8 = (chunk & 7) * 8;
            copy16_g2l(Bt + (long long)(tileN + row) * K + col8, &Blds[0][row * LDSTR + col8]);
        }
    }

    for (int i = 0; i < nk; i++) {
        wait_g2l();        // this wave's pending tile-i copies done
        __syncthreads();   // tile i visible; everyone done reading other buffer

        if (i + 1 < nk) {  // stage tile i+1 (overlaps compute below)
            int k0 = (i + 1) << 6;
            unsigned short* Ab = Alds[(i + 1) & 1];
            unsigned short* Bb = Blds[(i + 1) & 1];
            copy16_g2l(A + (long long)(tileM + arow0) * K + k0 + acol, Ab + arow0 * LDSTR + acol);
            copy16_g2l(A + (long long)(tileM + arow1) * K + k0 + acol, Ab + arow1 * LDSTR + acol);
            #pragma unroll
            for (int c = 0; c < 4; c++) {
                int chunk = tid + c * 256;
                int row = chunk >> 3, col8 = (chunk & 7) * 8;
                copy16_g2l(Bt + (long long)(tileN + row) * K + k0 + col8,
                           Bb + row * LDSTR + col8);
            }
        }

        const unsigned short* Ab = Alds[i & 1];
        const unsigned short* Bb = Blds[i & 1];
        #pragma unroll
        for (int ks = 0; ks < 2; ks++) {
            int kb = ks * 32;
            union { v16bf v; unsigned int u[8]; } a;
            const unsigned short* Ar = Ab + (waveM + l15) * LDSTR;
            #pragma unroll
            for (int e = 0; e < 8; e++) {
                int k = kb + ((e >> 2) << 4) + half * 8 + ((e & 3) << 1);
                a.u[e] = *(const unsigned int*)(Ar + k);
            }
            union { v16bf v; unsigned int u[8]; } bf[4];
            #pragma unroll
            for (int nt = 0; nt < 4; nt++) {
                const unsigned short* Br = Bb + (waveN + nt * 16 + l15) * LDSTR;
                #pragma unroll
                for (int e = 0; e < 8; e++) {
                    int k = kb + ((e >> 2) << 4) + half * 8 + ((e & 3) << 1);
                    bf[nt].u[e] = *(const unsigned int*)(Br + k);
                }
            }
            #pragma unroll
            for (int nt = 0; nt < 4; nt++) {
                acc[nt] = __builtin_amdgcn_wmma_f32_16x16x32_bf16(
                    false, a.v, false, bf[nt].v, (short)0, acc[nt], false, false);
            }
        }
    }

    #pragma unroll
    for (int nt = 0; nt < 4; nt++) {
        #pragma unroll
        for (int r = 0; r < 8; r++) {
            int m = tileM + waveM + half * 8 + r;
            int n = tileN + waveN + nt * 16 + l15;
            float val = acc[nt][r] + bias[n];
            long long o = (long long)m * N + n;
            if (residual) val += residual[o];
            if (relu) val = fmaxf(val, 0.f);
            if (outF) outF[o] = val;
            if (outB) outB[o] = f2bf(val);
        }
    }
}

// ---------------- v transpose: qkv_bf v-part -> vT[b,h,d,t] ------------------
__global__ void k_vT(const unsigned short* __restrict__ qkv,
                     unsigned short* __restrict__ vT) {
    int i = blockIdx.x * blockDim.x + threadIdx.x;   // < MM*CC
    int m = i / CC, c = i % CC;
    int b = m / TT, t = m % TT;
    int h = c >> 6, d = c & 63;
    vT[((long long)((b * HH + h) * DDIM + d)) * TT + t] =
        qkv[(long long)m * (3 * CC) + 2 * CC + c];
}

// ---------------- WMMA flash attention ---------------------------------------
// One wave per (b, h, 16-query tile); keys processed 32 at a time.
// S = Q K^T via 2 WMMAs / 16 keys; online softmax in C-layout (shfl_xor within
// 16-lane halves); P bounced through wave-private LDS into A-fragment layout;
// y += P V via 4 WMMAs / 32 keys using pre-transposed vT. All branches are
// wave-uniform so EXEC stays all-ones for every WMMA.
#define PSTR 40

__global__ void __launch_bounds__(256)
k_attn(const unsigned short* __restrict__ qkv,   // [M, 3C] bf16
       const unsigned short* __restrict__ vT,    // [B,H,D,T] bf16
       unsigned short* __restrict__ y) {         // [M, C] bf16
    __shared__ unsigned short Pst[8][16 * PSTR];

    int lane = threadIdx.x & 31;
    int wave = threadIdx.x >> 5;
    int l15  = lane & 15;
    int half = (lane >> 4) & 1;

    int gw = blockIdx.x * 8 + wave;               // global wave id, < B*H*(T/16)
    int nqt = TT / 16;
    int qt = gw % nqt;
    int h  = (gw / nqt) % HH;
    int b  = gw / (nqt * HH);
    int qbase = qt * 16;

    union AB { v16bf v; unsigned int u[8]; };
    v8f z = {0.f,0.f,0.f,0.f,0.f,0.f,0.f,0.f};

    // Q A-fragments: 16 queries x 64 dims, split into 2 k-steps of 32
    AB aQ[2];
    {
        const unsigned short* Qr =
            qkv + (long long)(b * TT + qbase + l15) * (3 * CC) + h * DDIM;
        #pragma unroll
        for (int ks = 0; ks < 2; ks++)
            #pragma unroll
            for (int e = 0; e < 8; e++) {
                int k = ks * 32 + ((e >> 2) << 4) + half * 8 + ((e & 3) << 1);
                aQ[ks].u[e] = *(const unsigned int*)(Qr + k);
            }
    }

    v8f yacc[4];
    #pragma unroll
    for (int i = 0; i < 4; i++) yacc[i] = z;
    float mrow[8], lrow[8];
    #pragma unroll
    for (int r = 0; r < 8; r++) { mrow[r] = -3.0e38f; lrow[r] = 0.f; }

    unsigned short* Pw = &Pst[wave][0];
    const unsigned short* Vbase =
        vT + (long long)((b * HH + h) * DDIM) * TT;

    int smax = qbase + 16;                        // exclusive key bound for tile
    for (int kb = 0; kb < smax; kb += 32) {
        // ---- scores for two 16-key blocks ----
        v8f S[2];
        #pragma unroll
        for (int blk2 = 0; blk2 < 2; blk2++) {
            int kb16 = kb + blk2 * 16;
            v8f s = z;
            if (kb16 < smax) {                    // wave-uniform
                const unsigned short* Kr =
                    qkv + (long long)(b * TT + kb16 + l15) * (3 * CC) + CC + h * DDIM;
                AB bK[2];
                #pragma unroll
                for (int ks = 0; ks < 2; ks++)
                    #pragma unroll
                    for (int e = 0; e < 8; e++) {
                        int d = ks * 32 + ((e >> 2) << 4) + half * 8 + ((e & 3) << 1);
                        bK[ks].u[e] = *(const unsigned int*)(Kr + d);
                    }
                s = __builtin_amdgcn_wmma_f32_16x16x32_bf16(
                    false, aQ[0].v, false, bK[0].v, (short)0, s, false, false);
                s = __builtin_amdgcn_wmma_f32_16x16x32_bf16(
                    false, aQ[1].v, false, bK[1].v, (short)0, s, false, false);
                #pragma unroll
                for (int r = 0; r < 8; r++) {
                    int q = qbase + r + half * 8;
                    s[r] = (kb16 + l15 <= q) ? s[r] * 0.125f : -3.0e38f;
                }
            } else {
                #pragma unroll
                for (int r = 0; r < 8; r++) s[r] = -3.0e38f;
            }
            S[blk2] = s;
        }

        // ---- online softmax across the 32-key chunk ----
        #pragma unroll
        for (int r = 0; r < 8; r++) {
            float mx = fmaxf(S[0][r], S[1][r]);
            #pragma unroll
            for (int off = 1; off < 16; off <<= 1)
                mx = fmaxf(mx, __shfl_xor(mx, off, 32));
            float mnew = fmaxf(mrow[r], mx);
            float corr = __expf(mrow[r] - mnew);
            mrow[r] = mnew;
            float p0 = __expf(S[0][r] - mnew);
            float p1 = __expf(S[1][r] - mnew);
            S[0][r] = p0; S[1][r] = p1;
            float ps = p0 + p1;
            #pragma unroll
            for (int off = 1; off < 16; off <<= 1)
                ps += __shfl_xor(ps, off, 32);
            lrow[r] = lrow[r] * corr + ps;
            #pragma unroll
            for (int nt = 0; nt < 4; nt++) yacc[nt][r] *= corr;
        }

        // ---- P (C-layout f32) -> LDS -> A-fragment (bf16) ----
        #pragma unroll
        for (int blk2 = 0; blk2 < 2; blk2++)
            #pragma unroll
            for (int r = 0; r < 8; r++)
                Pw[(r + half * 8) * PSTR + blk2 * 16 + l15] = f2bf(S[blk2][r]);

        AB aP;
        {
            const unsigned short* Pr = Pw + l15 * PSTR;
            #pragma unroll
            for (int e = 0; e < 8; e++) {
                int k = ((e >> 2) << 4) + half * 8 + ((e & 3) << 1);
                aP.u[e] = *(const unsigned int*)(Pr + k);
            }
        }

        // ---- y += P @ V (V from vT: key-pairs contiguous) ----
        #pragma unroll
        for (int nt = 0; nt < 4; nt++) {
            AB bV;
            const unsigned short* Vr = Vbase + (long long)(nt * 16 + l15) * TT + kb;
            #pragma unroll
            for (int e = 0; e < 8; e++) {
                int key = ((e >> 2) << 4) + half * 8 + ((e & 3) << 1);
                bV.u[e] = *(const unsigned int*)(Vr + key);
            }
            yacc[nt] = __builtin_amdgcn_wmma_f32_16x16x32_bf16(
                false, aP.v, false, bV.v, (short)0, yacc[nt], false, false);
        }
    }

    // ---- normalize and store ----
    #pragma unroll
    for (int r = 0; r < 8; r++) {
        float inv = 1.f / lrow[r];
        int m = b * TT + qbase + r + half * 8;
        #pragma unroll
        for (int nt = 0; nt < 4; nt++) {
            y[(long long)m * CC + h * DDIM + nt * 16 + l15] = f2bf(yacc[nt][r] * inv);
        }
    }
}

// =============================================================================
extern "C" void kernel_launch(void* const* d_in, const int* in_sizes, int n_in,
                              void* d_out, int out_size, void* d_ws, size_t ws_size,
                              hipStream_t stream) {
    const int*   idx   = (const int*)  d_in[0];
    const float* tok   = (const float*)d_in[1];
    const float* pos   = (const float*)d_in[2];
    const float* Wqkv  = (const float*)d_in[3];
    const float* bqkv  = (const float*)d_in[4];
    const float* Wproj = (const float*)d_in[5];
    const float* bproj = (const float*)d_in[6];
    const float* ln1g  = (const float*)d_in[7];
    const float* ln1b  = (const float*)d_in[8];
    const float* ln2g  = (const float*)d_in[9];
    const float* ln2b  = (const float*)d_in[10];
    const float* Wf1   = (const float*)d_in[11];
    const float* bf1   = (const float*)d_in[12];
    const float* Wf2   = (const float*)d_in[13];
    const float* bf2   = (const float*)d_in[14];
    const float* lnfg  = (const float*)d_in[15];
    const float* lnfb  = (const float*)d_in[16];
    const float* Whead = (const float*)d_in[17];
    const float* bhead = (const float*)d_in[18];
    float* logits = (float*)d_out;

    // ---- workspace carve (stream-ordered reuse across layers) ----
    char* p = (char*)d_ws;
    auto alloc = [&](size_t bytes) {
        char* r = p; p += (bytes + 255) & ~(size_t)255; return r;
    };
    unsigned short* wqkv_t  = (unsigned short*)alloc((size_t)3 * CC * CC * 2);
    unsigned short* wproj_t = (unsigned short*)alloc((size_t)CC * CC * 2);
    unsigned short* wf1_t   = (unsigned short*)alloc((size_t)4 * CC * CC * 2);
    unsigned short* wf2_t   = (unsigned short*)alloc((size_t)4 * CC * CC * 2);
    unsigned short* whead_t = (unsigned short*)alloc((size_t)VV * CC * 2);
    float*          x       = (float*)         alloc((size_t)MM * CC * 4);
    unsigned short* hbf     = (unsigned short*)alloc((size_t)MM * CC * 2);
    unsigned short* qkvbf   = (unsigned short*)alloc((size_t)MM * 3 * CC * 2);
    unsigned short* vT      = (unsigned short*)alloc((size_t)BB * HH * DDIM * TT * 2 + 256);
    unsigned short* ybf     = (unsigned short*)alloc((size_t)MM * CC * 2);
    unsigned short* abf     = (unsigned short*)alloc((size_t)MM * 4 * CC * 2);

    auto convT = [&](const float* in, unsigned short* out, int K, int N) {
        long long tot = (long long)K * N;
        int blocks = (int)((tot + 255) / 256);
        k_convT<<<blocks, 256, 0, stream>>>(in, out, K, N);
    };

    // embedding
    k_embed<<<(MM * CC) / 256, 256, 0, stream>>>(idx, tok, pos, x);

    dim3 blk(256);
    for (int l = 0; l < LL; l++) {
        convT(Wqkv  + (size_t)l * CC * 3 * CC, wqkv_t,  CC,     3 * CC);
        convT(Wproj + (size_t)l * CC * CC,     wproj_t, CC,     CC);
        convT(Wf1   + (size_t)l * CC * 4 * CC, wf1_t,   CC,     4 * CC);
        convT(Wf2   + (size_t)l * 4 * CC * CC, wf2_t,   4 * CC, CC);

        // ln1 -> bf16
        k_ln<<<MM, 256, 0, stream>>>(x, ln1g + l * CC, ln1b + l * CC, hbf);
        // qkv = h @ Wqkv + b   (bf16 out for WMMA attention)
        dim3 gq(3 * CC / 128, MM / 64);
        k_gemm<<<gq, blk, 0, stream>>>(hbf, wqkv_t, bqkv + (size_t)l * 3 * CC,
                                       nullptr, nullptr, qkvbf, MM, 3 * CC, CC, 0);
        // v transpose for P@V fragment layout
        k_vT<<<(MM * CC) / 256, 256, 0, stream>>>(qkvbf, vT);
        // WMMA flash attention -> bf16 y
        k_attn<<<(BB * HH * (TT / 16)) / 8, 256, 0, stream>>>(qkvbf, vT, ybf);
        // x = x + y @ Wproj + b
        dim3 gp(CC / 128, MM / 64);
        k_gemm<<<gp, blk, 0, stream>>>(ybf, wproj_t, bproj + (size_t)l * CC,
                                       x, x, nullptr, MM, CC, CC, 0);
        // ln2 -> bf16
        k_ln<<<MM, 256, 0, stream>>>(x, ln2g + l * CC, ln2b + l * CC, hbf);
        // a = relu(h2 @ Wf1 + b) -> bf16
        dim3 gf(4 * CC / 128, MM / 64);
        k_gemm<<<gf, blk, 0, stream>>>(hbf, wf1_t, bf1 + (size_t)l * 4 * CC,
                                       nullptr, nullptr, abf, MM, 4 * CC, CC, 1);
        // x = x + a @ Wf2 + b
        k_gemm<<<gp, blk, 0, stream>>>(abf, wf2_t, bf2 + (size_t)l * CC,
                                       x, x, nullptr, MM, CC, 4 * CC, 0);
    }

    // final LN + head GEMM -> logits fp32
    k_ln<<<MM, 256, 0, stream>>>(x, lnfg, lnfb, hbf);
    convT(Whead, whead_t, CC, VV);
    dim3 gh(VV / 128, MM / 64);
    k_gemm<<<gh, blk, 0, stream>>>(hbf, whead_t, bhead,
                                   nullptr, logits, nullptr, MM, VV, CC, 0);
}
